// FastAttention_78271484003204
// MI455X (gfx1250) — compile-verified, matrix operand order checked by reading, not run on hardware
//
#include <hip/hip_runtime.h>
#include <hip/hip_bf16.h>

typedef __attribute__((ext_vector_type(16))) _Float16 v16h;
typedef __attribute__((ext_vector_type(8)))  _Float16 v8h;
typedef __attribute__((ext_vector_type(8)))  float    v8f;

namespace {
constexpr int Bc = 4, Hc = 16, Nc = 4096, Dc = 64, Mc = 266, Ec = 64;
constexpr int MW  = 272;   // feature rows kept in workspace (multiple of 16)
constexpr int MT  = 17;    // 272/16 m-tiles
constexpr int CW  = 80;    // ws context row stride (64 v-cols + ksum col + pad)
constexpr float DN    = 0.35355339059327379f;  // 64^-0.25
constexpr float RATIO = 0.06131393394849658f;  // 266^-0.5
constexpr float EPSV  = 1e-4f;
}

// ---------- helpers ----------
__device__ inline unsigned fenc(float x) {
  unsigned u = __float_as_uint(x);
  return (u & 0x80000000u) ? ~u : (u | 0x80000000u);
}
__device__ inline float fdec(unsigned e) {
  return __uint_as_float((e & 0x80000000u) ? (e & 0x7fffffffu) : ~e);
}
// K index held by (lane, half-slot j) in a swizzled 16x32 operand tile
__device__ inline int swz_K(int ln, int j) {
  int kb = (ln & 16) ? 8 : 0;
  return (j < 8) ? (kb + j) : (16 + kb + (j - 8));
}
// one operand tile = 512 halves: lane*16 + j
__device__ inline v16h load_T(const _Float16* buf, int tile, int lane) {
  return *(const v16h*)(buf + (tile << 9) + (lane << 4));
}
__device__ inline v8f wmma16(v16h a, v16h b, v8f c) {
  return __builtin_amdgcn_wmma_f32_16x16x32_f16(false, a, false, b, (short)0, c,
                                                false, false);
}

// ws layout: [0] encoded global kmax; [16..) context_ext [64][MW][CW] (col 64 = ksum)
__device__ __host__ inline size_t ws_ctx_off() { return 16; }

// ---------- kernel 0: init workspace ----------
__global__ void init_ws(float* ws, int nfloat) {
  int i = blockIdx.x * blockDim.x + threadIdx.x;
  if (i == 0) ((unsigned*)ws)[0] = fenc(-__builtin_inff());
  for (int j = i + 16; j < nfloat; j += gridDim.x * blockDim.x) ws[j] = 0.f;
}

// ---------- kernel 1: global max of k_dash ----------
__global__ __launch_bounds__(256) void kmax_kernel(const float* __restrict__ k,
                                                   const float* __restrict__ proj,
                                                   float* ws) {
  extern __shared__ char smem[];
  _Float16* sProjB = (_Float16*)smem;           // 34 tiles: (mt,kc) B-layout
  _Float16* sKA    = sProjB + 34 * 512;         // 16 tiles: (nt,kc) A-layout
  unsigned* sWMax  = (unsigned*)(sKA + 16 * 512);
  const int tid = threadIdx.x, lane = tid & 31, wave = tid >> 5;
  const int bh = blockIdx.x >> 5, slab = blockIdx.x & 31;
  const float* kb = k + ((size_t)bh * Nc + (size_t)slab * 128) * Dc;

  for (int idx = tid; idx < 34 * 512; idx += 256) {
    int tile = idx >> 9, r = idx & 511, ln = r >> 4, j = r & 15;
    int m = (tile >> 1) * 16 + (ln & 15);
    int d = (tile & 1) * 32 + swz_K(ln, j);
    sProjB[idx] = (m < Mc) ? (_Float16)proj[m * Dc + d] : (_Float16)0.f;
  }
  for (int idx = tid; idx < 16 * 512; idx += 256) {
    int tile = idx >> 9, r = idx & 511, ln = r >> 4, j = r & 15;
    int n = (tile >> 1) * 16 + (ln & 15);
    int d = (tile & 1) * 32 + swz_K(ln, j);
    sKA[idx] = (_Float16)(kb[n * Dc + d] * DN);
  }
  __syncthreads();

  float wmax = -__builtin_inff();
  for (int mt = 0; mt < MT; ++mt) {
    v8f acc = {};
    acc = wmma16(load_T(sKA, wave * 2 + 0, lane), load_T(sProjB, mt * 2 + 0, lane), acc);
    acc = wmma16(load_T(sKA, wave * 2 + 1, lane), load_T(sProjB, mt * 2 + 1, lane), acc);
    int m = mt * 16 + (lane & 15);
    if (m < Mc) {
#pragma unroll
      for (int r = 0; r < 8; ++r) wmax = fmaxf(wmax, acc[r]);
    }
  }
  for (int off = 16; off; off >>= 1) wmax = fmaxf(wmax, __shfl_xor(wmax, off, 32));
  if (lane == 0) sWMax[wave] = fenc(wmax);
  __syncthreads();
  if (tid == 0) {
    unsigned e = sWMax[0];
#pragma unroll
    for (int w = 1; w < 8; ++w) e = max(e, sWMax[w]);
    atomicMax((unsigned*)ws, e);
  }
}

// ---------- kernel 2: kp -> context_ext (incl. ksum column) ----------
__global__ __launch_bounds__(256) void ctx_kernel(const float* __restrict__ k,
                                                  const float* __restrict__ v,
                                                  const float* __restrict__ proj,
                                                  float* ws) {
  extern __shared__ char smem[];
  _Float16* sProjB = (_Float16*)smem;            // 34 tiles (mt,kc)
  _Float16* sKA    = sProjB + 34 * 512;          // 16 tiles (nt,kc)
  _Float16* sVB    = sKA + 16 * 512;             // 20 tiles (kc,et) et<5, col64=ones
  _Float16* sKPA   = sVB + 20 * 512;             // 68 tiles (mt,kcn): kp^T A-layout
  float*    sDiag  = (float*)(sKPA + 68 * 512);  // [128]
  const int tid = threadIdx.x, lane = tid & 31, wave = tid >> 5;
  const int bh = blockIdx.x >> 3, part = blockIdx.x & 7;
  const float* kbp = k + ((size_t)bh * Nc + (size_t)part * 512) * Dc;
  const float* vbp = v + ((size_t)bh * Nc + (size_t)part * 512) * Dc;
  float* ctx = ws + ws_ctx_off() + (size_t)bh * MW * CW;
  const float gmax = fdec(((const unsigned*)ws)[0]);

  for (int idx = tid; idx < 34 * 512; idx += 256) {
    int tile = idx >> 9, r = idx & 511, ln = r >> 4, j = r & 15;
    int m = (tile >> 1) * 16 + (ln & 15);
    int d = (tile & 1) * 32 + swz_K(ln, j);
    sProjB[idx] = (m < Mc) ? (_Float16)proj[m * Dc + d] : (_Float16)0.f;
  }

  v8f acc[11];
#pragma unroll
  for (int i = 0; i < 11; ++i) acc[i] = v8f{};

  for (int s = 0; s < 4; ++s) {
    const float* kslab = kbp + (size_t)s * 128 * Dc;
    const float* vslab = vbp + (size_t)s * 128 * Dc;
    __syncthreads();
    for (int idx = tid; idx < 16 * 512; idx += 256) {
      int tile = idx >> 9, r = idx & 511, ln = r >> 4, j = r & 15;
      int n = (tile >> 1) * 16 + (ln & 15);
      int d = (tile & 1) * 32 + swz_K(ln, j);
      sKA[idx] = (_Float16)(kslab[n * Dc + d] * DN);
    }
    for (int idx = tid; idx < 20 * 512; idx += 256) {
      int tile = idx >> 9, r = idx & 511, ln = r >> 4, j = r & 15;
      int kc = tile / 5, et = tile - kc * 5;
      int e = et * 16 + (ln & 15);
      int n = kc * 32 + swz_K(ln, j);
      float val = (e < Ec) ? vslab[n * Dc + e] : (e == Ec ? 1.f : 0.f);
      sVB[idx] = (_Float16)val;
    }
    if (tid < 128) {
      float ssq = 0.f;
      const float* kr = kslab + tid * Dc;
#pragma unroll 8
      for (int d = 0; d < Dc; ++d) { float x = kr[d]; ssq += x * x; }
      sDiag[tid] = ssq * 0.0625f;  // * dn^2 / 2
    }
    __syncthreads();

    // stage 1: k_dash tiles, stored transposed into A-layout (contiguous v8h)
    const int kcn = wave >> 1, jb = (wave & 1) * 8;
    for (int mt = 0; mt < MT; ++mt) {
      v8f d = {};
      d = wmma16(load_T(sKA, wave * 2 + 0, lane), load_T(sProjB, mt * 2 + 0, lane), d);
      d = wmma16(load_T(sKA, wave * 2 + 1, lane), load_T(sProjB, mt * 2 + 1, lane), d);
      v8h h;
#pragma unroll
      for (int r = 0; r < 8; ++r) h[r] = (_Float16)d[r];
      *(v8h*)(sKPA + ((mt * 4 + kcn) << 9) + (lane << 4) + jb) = h;
    }
    __syncthreads();

    // stage 1b: exp -> kp (vectorized in-place)
    for (int q2 = tid; q2 < 68 * 32; q2 += 256) {
      int tile = q2 >> 5, ln = q2 & 31;
      int mt = tile >> 2, kcn2 = tile & 3;
      int m = mt * 16 + (ln & 15);
      _Float16* p = sKPA + (tile << 9) + (ln << 4);
      v16h y = {};
      if (m < Mc) {
        v16h x = *(const v16h*)p;
#pragma unroll
        for (int j = 0; j < 16; ++j) {
          int n = kcn2 * 32 + swz_K(ln, j);
          y[j] = (_Float16)(RATIO * (__expf((float)x[j] - sDiag[n] - gmax) + EPSV));
        }
      }
      *(v16h*)p = y;
    }
    __syncthreads();

    // stage 2: context_ext += kp^T @ [v | 1]
    int ti = 0;
    for (int t = wave; t < MT * 5; t += 8, ++ti) {
      int mt = t / 5, et = t - mt * 5;
      v8f c = acc[ti];
#pragma unroll
      for (int kc = 0; kc < 4; ++kc)
        c = wmma16(load_T(sKPA, mt * 4 + kc, lane), load_T(sVB, kc * 5 + et, lane), c);
      acc[ti] = c;
    }
  }

  int ti = 0;
  for (int t = wave; t < MT * 5; t += 8, ++ti) {
    int mt = t / 5, et = t - mt * 5;
    int e = et * 16 + (lane & 15);
    if (e <= Ec) {
#pragma unroll
      for (int r = 0; r < 8; ++r) {
        int m = mt * 16 + ((lane < 16) ? r : r + 8);
        atomicAdd(&ctx[(size_t)m * CW + e], acc[ti][r]);
      }
    }
  }
}

// ---------- kernel 3: qp, normalizer (via ksum column), out ----------
__global__ __launch_bounds__(256) void out_kernel(const float* __restrict__ q,
                                                  const float* __restrict__ proj,
                                                  const float* __restrict__ ws,
                                                  float* __restrict__ out) {
  extern __shared__ char smem[];
  // union buffer: stage1 = projA(34 tiles) + qB(16 tiles); stage2 = ctxB(45 tiles)
  _Float16* sBB   = (_Float16*)smem;            // 50 tiles
  _Float16* sProjA = sBB;                       // tiles 0..33 (A-layout, rows=m)
  _Float16* sQB    = sBB + 34 * 512;            // tiles (kc,nt) B-layout (q^T)
  _Float16* sQPA  = sBB + 50 * 512;             // 72 tiles (nt,kcm): qp A-layout
  float*    sDiag = (float*)(sQPA + 72 * 512);  // [128]
  float*    sRMax = sDiag + 128;                // [128]
  float*    sDinv = sRMax + 128;                // [128]
  const int tid = threadIdx.x, lane = tid & 31, wave = tid >> 5;
  const int bh = blockIdx.x >> 5, slab = blockIdx.x & 31;
  const float* qb = q + ((size_t)bh * Nc + (size_t)slab * 128) * Dc;
  const float* ctx = ws + ws_ctx_off() + (size_t)bh * MW * CW;

  for (int idx = tid; idx < 34 * 512; idx += 256) {
    int tile = idx >> 9, r = idx & 511, ln = r >> 4, j = r & 15;
    int m = (tile >> 1) * 16 + (ln & 15);
    int d = (tile & 1) * 32 + swz_K(ln, j);
    sProjA[idx] = (m < Mc) ? (_Float16)proj[m * Dc + d] : (_Float16)0.f;
  }
  for (int idx = tid; idx < 16 * 512; idx += 256) {
    int tile = idx >> 9, r = idx & 511, ln = r >> 4, j = r & 15;
    int kc = tile >> 3, nt = tile & 7;
    int n = nt * 16 + (ln & 15);
    int d = kc * 32 + swz_K(ln, j);
    sQB[idx] = (_Float16)(qb[n * Dc + d] * DN);
  }
  if (tid < 128) {
    float ssq = 0.f;
    const float* qr = qb + tid * Dc;
#pragma unroll 8
    for (int d = 0; d < Dc; ++d) { float x = qr[d]; ssq += x * x; }
    sDiag[tid] = ssq * 0.0625f;
  }
  __syncthreads();

  // stage 1: dash^T = proj @ q^T  (C rows = m) -> contiguous store into qp A-layout
  // wave w owns columns n = w*16 .. w*16+15  -> private row-max, no atomics
  float rmax = -__builtin_inff();
  for (int mt = 0; mt < MT; ++mt) {
    v8f d = {};
    d = wmma16(load_T(sProjA, mt * 2 + 0, lane), load_T(sQB, 0 * 8 + wave, lane), d);
    d = wmma16(load_T(sProjA, mt * 2 + 1, lane), load_T(sQB, 1 * 8 + wave, lane), d);
    v8h h;
#pragma unroll
    for (int r = 0; r < 8; ++r) {
      h[r] = (_Float16)d[r];
      int m = mt * 16 + ((lane < 16) ? r : r + 8);
      if (m < Mc) rmax = fmaxf(rmax, d[r]);
    }
    *(v8h*)(sQPA + ((wave * 9 + (mt >> 1)) << 9) + (lane << 4) + (mt & 1) * 8) = h;
  }
  rmax = fmaxf(rmax, __shfl_xor(rmax, 16, 32));
  if (lane < 16) sRMax[wave * 16 + lane] = rmax;
  __syncthreads();

  // stage 1b: exp -> qp (vectorized); also swap sBB to context_ext (f16, B-layout)
  for (int q2 = tid; q2 < 72 * 32; q2 += 256) {
    int tile = q2 >> 5, ln = q2 & 31;
    int nt = tile / 9, kcm = tile - nt * 9;
    int n = nt * 16 + (ln & 15);
    float dg = sDiag[n], rm = sRMax[n];
    _Float16* p = sQPA + (tile << 9) + (ln << 4);
    v16h x = *(const v16h*)p;
    v16h y = {};
#pragma unroll
    for (int j = 0; j < 16; ++j) {
      int m = kcm * 32 + swz_K(ln, j);
      if (m < Mc)
        y[j] = (_Float16)(RATIO * (__expf((float)x[j] - dg - rm) + EPSV));
    }
    *(v16h*)p = y;
  }
  for (int idx = tid; idx < 45 * 512; idx += 256) {
    int tile = idx >> 9, r = idx & 511, ln = r >> 4, j = r & 15;
    int kc = tile / 5, et = tile - kc * 5;
    int e = et * 16 + (ln & 15);
    int m = kc * 32 + swz_K(ln, j);
    float val = (m < MW && e <= Ec) ? ctx[(size_t)m * CW + e] : 0.f;
    sBB[idx] = (_Float16)val;
  }
  __syncthreads();

  // stage 2: [out | d] = qp @ ctx_ext ; col 64 gives normalizer
  v8f acc[5];
#pragma unroll
  for (int i = 0; i < 5; ++i) {
    int t = wave + 8 * i;
    int nt = t / 5, et = t - nt * 5;
    v8f c = {};
#pragma unroll
    for (int kc = 0; kc < 9; ++kc)
      c = wmma16(load_T(sQPA, nt * 9 + kc, lane), load_T(sBB, kc * 5 + et, lane), c);
    acc[i] = c;
    if (et == 4 && (lane & 15) == 0) {  // e == 64 column
#pragma unroll
      for (int r = 0; r < 8; ++r) {
        int row = nt * 16 + ((lane < 16) ? r : r + 8);
        sDinv[row] = 1.f / c[r];
      }
    }
  }
  __syncthreads();

  float* ob = out + ((size_t)bh * Nc + (size_t)slab * 128) * Ec;
#pragma unroll
  for (int i = 0; i < 5; ++i) {
    int t = wave + 8 * i;
    int nt = t / 5, et = t - nt * 5;
    if (et < 4) {
      int e = et * 16 + (lane & 15);
#pragma unroll
      for (int r = 0; r < 8; ++r) {
        int row = nt * 16 + ((lane < 16) ? r : r + 8);
        ob[(size_t)row * Ec + e] = acc[i][r] * sDinv[row];
      }
    }
  }
}

// ---------- host launcher ----------
extern "C" void kernel_launch(void* const* d_in, const int* in_sizes, int n_in,
                              void* d_out, int out_size, void* d_ws, size_t ws_size,
                              hipStream_t stream) {
  const float* q    = (const float*)d_in[0];
  const float* k    = (const float*)d_in[1];
  const float* v    = (const float*)d_in[2];
  const float* proj = (const float*)d_in[3];
  float* out = (float*)d_out;
  float* ws  = (float*)d_ws;
  (void)in_sizes; (void)n_in; (void)out_size; (void)ws_size;

  const int nfloat = (int)(16 + (size_t)Bc * Hc * MW * CW);
  init_ws<<<512, 256, 0, stream>>>(ws, nfloat);

  const size_t smemA = (size_t)(34 + 16) * 512 * sizeof(_Float16) + 8 * sizeof(unsigned);
  kmax_kernel<<<dim3(Bc * Hc * (Nc / 128)), 256, smemA, stream>>>(k, proj, ws);

  const size_t smemB = (size_t)(34 + 16 + 20 + 68) * 512 * sizeof(_Float16)
                       + 128 * sizeof(float);
  ctx_kernel<<<dim3(Bc * Hc * 8), 256, smemB, stream>>>(k, v, proj, ws);

  const size_t smemC = (size_t)(50 + 72) * 512 * sizeof(_Float16)
                       + (size_t)(128 + 128 + 128) * sizeof(float);
  out_kernel<<<dim3(Bc * Hc * (Nc / 128)), 256, smemC, stream>>>(q, proj, ws, out);
}
